// ANI2x_7988639171028
// MI455X (gfx1250) — compile-verified, matrix operand order checked by reading, not run on hardware
//
#include <hip/hip_runtime.h>

// ---------------------------------------------------------------------------
// ANI-2x energy + forces for MI455X (gfx1250, wave32, WMMA).
// GEMMs use v_wmma_f32_16x16x32_f16 (f32 accumulate) with 16x(16*NT) register
// blocking per wave (A-fragment reuse across NT WMMAs). AEV build/backprop are
// scalar VALU kernels with atomic scatter. All scratch in d_ws.
// ---------------------------------------------------------------------------

typedef _Float16 h8   __attribute__((ext_vector_type(8)));
typedef _Float16 v16h __attribute__((ext_vector_type(16)));
typedef float    v8f  __attribute__((ext_vector_type(8)));

#define DEVINL __device__ __forceinline__

constexpr int   NA   = 2100;   // atoms
constexpr int   SPN  = 7;      // species
constexpr int   EN   = 8;      // ensemble
constexpr int   MS   = 300;    // atoms per species
constexpr int   MP   = 304;    // padded to 19 * 16
constexpr int   MT   = 19;     // M tiles
constexpr int   AEVD = 1008;   // AEV length (7*16 + 28*32)
constexpr int   K0   = 1024;   // padded AEV for GEMM
constexpr int   N1   = 256;
constexpr int   N2   = 192;
constexpr int   N3   = 160;
constexpr float RCR_ = 5.1f;
constexpr float RCA_ = 3.5f;
constexpr float ETAR = 19.7f;
constexpr float ETAA = 12.5f;
constexpr float ZET  = 14.1f;
constexpr float PIF  = 3.14159265358979323846f;

DEVINL float fcut (float d, float rc) { return 0.5f * __cosf(PIF * d / rc) + 0.5f; }
DEVINL float fcutp(float d, float rc) { return -0.5f * (PIF / rc) * __sinf(PIF * d / rc); }

// ---------------------------------------------------------------------------
// utility kernels
// ---------------------------------------------------------------------------
__global__ void k_zero(float* __restrict__ p, long n) {
    long i = (long)blockIdx.x * blockDim.x + threadIdx.x;
    if (i < n) p[i] = 0.0f;
}

// Convert f32 weights [SE][K][N] into f16 Wh [SE][Kpad][N] and Wt [SE][N][Kpad]
__global__ void k_convw(const float* __restrict__ src, _Float16* __restrict__ wh,
                        _Float16* __restrict__ wt, int K, int Kpad, int N, long total) {
    long t = (long)blockIdx.x * blockDim.x + threadIdx.x;
    if (t >= total) return;
    int n  = (int)(t % N);
    long r = t / N;
    int k  = (int)(r % Kpad);
    long se = r / Kpad;
    float v = (k < K) ? src[(se * K + k) * N + n] : 0.0f;
    wh[(se * Kpad + k) * (long)N + n] = (_Float16)v;
    wt[(se * N + n) * (long)Kpad + k] = (_Float16)v;
}

// ---------------------------------------------------------------------------
// AEV forward
// ---------------------------------------------------------------------------
__global__ void k_rad_fwd(const float* __restrict__ pos, const int* __restrict__ sp,
                          const int* __restrict__ i12, int P, float* __restrict__ aev) {
    int p = blockIdx.x * blockDim.x + threadIdx.x;
    if (p >= P) return;
    int i = i12[p], j = i12[P + p];
    float dx = pos[3 * j] - pos[3 * i];
    float dy = pos[3 * j + 1] - pos[3 * i + 1];
    float dz = pos[3 * j + 2] - pos[3 * i + 2];
    float d  = sqrtf(dx * dx + dy * dy + dz * dz);
    float fc = fcut(d, RCR_);
    int si = sp[i], sj = sp[j];
    #pragma unroll
    for (int t = 0; t < 16; ++t) {
        float df = d - (0.8f + 0.26875f * t);
        float r  = 0.25f * __expf(-ETAR * df * df) * fc;
        atomicAdd(&aev[(long)i * AEVD + sj * 16 + t], r);
        atomicAdd(&aev[(long)j * AEVD + si * 16 + t], r);
    }
}

__global__ void k_ang_fwd(const float* __restrict__ pos, const int* __restrict__ sp,
                          const int* __restrict__ tc, const int* __restrict__ tj,
                          const int* __restrict__ tk, int T, float* __restrict__ aev) {
    int t = blockIdx.x * blockDim.x + threadIdx.x;
    if (t >= T) return;
    int c = tc[t], j = tj[t], k = tk[t];
    float vjx = pos[3 * j] - pos[3 * c], vjy = pos[3 * j + 1] - pos[3 * c + 1], vjz = pos[3 * j + 2] - pos[3 * c + 2];
    float vkx = pos[3 * k] - pos[3 * c], vky = pos[3 * k + 1] - pos[3 * c + 1], vkz = pos[3 * k + 2] - pos[3 * c + 2];
    float dj = sqrtf(vjx * vjx + vjy * vjy + vjz * vjz);
    float dk = sqrtf(vkx * vkx + vky * vky + vkz * vkz);
    float u  = (vjx * vkx + vjy * vky + vjz * vkz) / (dj * dk);
    u = fminf(1.0f, fmaxf(-1.0f, u));
    float th = acosf(0.95f * u);
    float fc = fcut(dj, RCA_) * fcut(dk, RCA_);
    float davg = 0.5f * (dj + dk);
    int sj_ = sp[j], sk_ = sp[k];
    int s1 = min(sj_, sk_), s2 = max(sj_, sk_);
    int pidx = s1 * SPN - (s1 * (s1 + 1)) / 2 + s2;
    float* dst = &aev[(long)c * AEVD + 112 + pidx * 32];
    float f2[8];
    #pragma unroll
    for (int a = 0; a < 8; ++a) {
        float q = davg - (0.8f + 0.3375f * a);
        f2[a] = __expf(-ETAA * q * q);
    }
    #pragma unroll
    for (int z = 0; z < 4; ++z) {
        float phi = th - (z + 0.5f) * (PIF * 0.25f);
        float b   = 0.5f * (1.0f + __cosf(phi));
        float f1  = __powf(b, ZET);
        #pragma unroll
        for (int a = 0; a < 8; ++a)
            atomicAdd(&dst[z * 8 + a], 2.0f * f1 * f2[a] * fc);
    }
}

// gather species rows of AEV into padded f16 X [SPN][MP][K0]
__global__ void k_gather_x(const float* __restrict__ aev, const int* __restrict__ idx,
                           _Float16* __restrict__ xh, long total) {
    long t = (long)blockIdx.x * blockDim.x + threadIdx.x;
    if (t >= total) return;
    int k  = (int)(t & (K0 - 1));
    long r = t >> 10;
    int m  = (int)(r % MP);
    int s  = (int)(r / MP);
    float v = 0.0f;
    if (m < MS && k < AEVD) v = aev[(long)idx[s * MS + m] * AEVD + k];
    xh[t] = (_Float16)v;
}

// ---------------------------------------------------------------------------
// WMMA GEMM: each wave32 computes a 16 x (16*NT) output tile. The A fragment
// is loaded once per K-chunk and reused across NT WMMAs (register blocking).
// MODE 0 (fwd):  Oh = celu(A@B + bias), Od = celu'(pre)      (f16, f16)
// MODE 1 (bwd):  Oh = (A@B) * Dm                              (f16)
// MODE 2 (bwdX): Of = sum_e (A_e @ B_e)                       (f32)
// A row-major [M][K]; B given as row-major [Ncols][K] (transposed weight copy)
// so both fragments use identical contiguous b128 loads.
// ---------------------------------------------------------------------------
DEVINL v16h frag16(const _Float16* __restrict__ p) {
    h8 lo = *(const h8*)p;
    h8 hi = *(const h8*)(p + 16);
    v16h r;
    #pragma unroll
    for (int j = 0; j < 8; ++j) { r[j] = lo[j]; r[j + 8] = hi[j]; }
    return r;
}

template <int MODE, int NT>
__global__ __launch_bounds__(32) void k_gemm(
    const _Float16* __restrict__ A, long sAs, long sAe, int lda,
    const _Float16* __restrict__ B, long sBs, long sBe, int ldb,
    const float* __restrict__ bias, int nbias,
    const _Float16* __restrict__ Dm, long sDs, long sDe,
    _Float16* __restrict__ Oh, long sOs, long sOe, int ldo,
    _Float16* __restrict__ Od,
    float* __restrict__ Of, long sFs,
    int K, int NtilesW, int Eloop) {
    int lane = threadIdx.x;
    int s = blockIdx.z, e = blockIdx.y;
    int mt = blockIdx.x / NtilesW, nt = blockIdx.x % NtilesW;
    int half = lane >> 4, l15 = lane & 15;
    int m0 = mt * 16, n0 = nt * (16 * NT);
    v8f acc[NT];
    #pragma unroll
    for (int q = 0; q < NT; ++q) acc[q] = {0.f, 0.f, 0.f, 0.f, 0.f, 0.f, 0.f, 0.f};
    for (int el = 0; el < Eloop; ++el) {
        const _Float16* arow = A + (long)s * sAs + (long)(e + el) * sAe + (long)(m0 + l15) * lda + half * 8;
        const _Float16* brow = B + (long)s * sBs + (long)(e + el) * sBe + (long)(n0 + l15) * ldb + half * 8;
        for (int kb = 0; kb < K; kb += 32) {
            if (kb + 128 < K) {  // global_prefetch_b8 of upcoming weight chunk
                __builtin_prefetch(brow + kb + 128, 0, 1);
            }
            v16h av = frag16(arow + kb);
            #pragma unroll
            for (int q = 0; q < NT; ++q) {
                v16h bv = frag16(brow + (long)q * 16 * ldb + kb);
                acc[q] = __builtin_amdgcn_wmma_f32_16x16x32_f16(false, av, false, bv,
                                                                (short)0, acc[q], false, false);
            }
        }
    }
    int mbase = m0 + 8 * half;
    if (MODE == 0) {
        const float* bp = bias + ((long)s * EN + e) * nbias;
        _Float16* oh = Oh + (long)s * sOs + (long)e * sOe;
        _Float16* od = Od + (long)s * sOs + (long)e * sOe;
        #pragma unroll
        for (int q = 0; q < NT; ++q) {
            int n = n0 + q * 16 + l15;
            float bv = bp[n];
            #pragma unroll
            for (int r = 0; r < 8; ++r) {
                int m = mbase + r;
                float pre = acc[q][r] + bv;
                float val, der;
                if (pre > 0.0f) { val = pre; der = 1.0f; }
                else { float tt = __expf(pre * 10.0f); val = 0.1f * (tt - 1.0f); der = tt; }
                oh[(long)m * ldo + n] = (_Float16)val;
                od[(long)m * ldo + n] = (_Float16)der;
            }
        }
    } else if (MODE == 1) {
        const _Float16* dm = Dm + (long)s * sDs + (long)e * sDe;
        _Float16* oh = Oh + (long)s * sOs + (long)e * sOe;
        #pragma unroll
        for (int q = 0; q < NT; ++q) {
            int n = n0 + q * 16 + l15;
            #pragma unroll
            for (int r = 0; r < 8; ++r) {
                int m = mbase + r;
                float g = acc[q][r] * (float)dm[(long)m * ldo + n];
                oh[(long)m * ldo + n] = (_Float16)g;
            }
        }
    } else {
        float* of = Of + (long)s * sFs;
        #pragma unroll
        for (int q = 0; q < NT; ++q) {
            int n = n0 + q * 16 + l15;
            #pragma unroll
            for (int r = 0; r < 8; ++r)
                of[(long)(mbase + r) * ldo + n] = acc[q][r];
        }
    }
}

// ---------------------------------------------------------------------------
// layer 3 (160 -> 1) + energy reduction; and G2 seed for backward
// ---------------------------------------------------------------------------
__global__ void k_l3_energy(const _Float16* __restrict__ h2, const float* __restrict__ W3,
                            const float* __restrict__ b3, const float* __restrict__ sae,
                            float* __restrict__ en) {
    long t = (long)blockIdx.x * blockDim.x + threadIdx.x;
    if (t >= (long)SPN * EN * MS) return;
    int m  = (int)(t % MS);
    long r = t / MS;
    int e  = (int)(r % EN);
    int s  = (int)(r / EN);
    long se = (long)s * EN + e;
    const _Float16* hp = h2 + (se * MP + m) * (long)N3;
    const float* wp = W3 + se * N3;
    float acc = b3[se];
    #pragma unroll 8
    for (int k = 0; k < N3; ++k) acc += (float)hp[k] * wp[k];
    atomicAdd(en, 0.125f * acc);
    if (e == 0 && m == 0) atomicAdd(en, sae[s] * (float)MS);
}

__global__ void k_g2_seed(const _Float16* __restrict__ d2, const float* __restrict__ W3,
                          _Float16* __restrict__ g2, long total) {
    long t = (long)blockIdx.x * blockDim.x + threadIdx.x;
    if (t >= total) return;
    int n  = (int)(t % N3);
    long r = t / N3;
    int m  = (int)(r % MP);
    long se = r / MP;
    float v = 0.0f;
    if (m < MS) v = 0.125f * W3[se * N3 + n] * (float)d2[t];
    g2[t] = (_Float16)v;
}

// scatter dX back into dAEV per-atom rows (each atom belongs to one species)
__global__ void k_scatter_daev(const float* __restrict__ gx, const int* __restrict__ idx,
                               float* __restrict__ daev, long total) {
    long t = (long)blockIdx.x * blockDim.x + threadIdx.x;
    if (t >= total) return;
    int k  = (int)(t % AEVD);
    long r = t / AEVD;
    int m  = (int)(r % MS);
    int s  = (int)(r / MS);
    daev[(long)idx[s * MS + m] * AEVD + k] = gx[((long)s * MP + m) * K0 + k];
}

// ---------------------------------------------------------------------------
// AEV backward -> coordinate gradients
// ---------------------------------------------------------------------------
__global__ void k_rad_bwd(const float* __restrict__ pos, const int* __restrict__ sp,
                          const int* __restrict__ i12, int P,
                          const float* __restrict__ daev, float* __restrict__ grad) {
    int p = blockIdx.x * blockDim.x + threadIdx.x;
    if (p >= P) return;
    int i = i12[p], j = i12[P + p];
    float dx = pos[3 * j] - pos[3 * i];
    float dy = pos[3 * j + 1] - pos[3 * i + 1];
    float dz = pos[3 * j + 2] - pos[3 * i + 2];
    float d  = sqrtf(dx * dx + dy * dy + dz * dz);
    float fc = fcut(d, RCR_);
    float fp = fcutp(d, RCR_);
    int si = sp[i], sj = sp[j];
    float gsum = 0.0f;
    #pragma unroll
    for (int t = 0; t < 16; ++t) {
        float g  = daev[(long)i * AEVD + sj * 16 + t] + daev[(long)j * AEVD + si * 16 + t];
        float df = d - (0.8f + 0.26875f * t);
        float ex = __expf(-ETAR * df * df);
        gsum += g * 0.25f * (ex * (-2.0f * ETAR * df) * fc + ex * fp);
    }
    float c = gsum / d;
    atomicAdd(&grad[3 * j + 0],  c * dx);
    atomicAdd(&grad[3 * j + 1],  c * dy);
    atomicAdd(&grad[3 * j + 2],  c * dz);
    atomicAdd(&grad[3 * i + 0], -c * dx);
    atomicAdd(&grad[3 * i + 1], -c * dy);
    atomicAdd(&grad[3 * i + 2], -c * dz);
}

__global__ void k_ang_bwd(const float* __restrict__ pos, const int* __restrict__ sp,
                          const int* __restrict__ tc, const int* __restrict__ tj,
                          const int* __restrict__ tk, int T,
                          const float* __restrict__ daev, float* __restrict__ grad) {
    int t = blockIdx.x * blockDim.x + threadIdx.x;
    if (t >= T) return;
    int c = tc[t], j = tj[t], k = tk[t];
    float vjx = pos[3 * j] - pos[3 * c], vjy = pos[3 * j + 1] - pos[3 * c + 1], vjz = pos[3 * j + 2] - pos[3 * c + 2];
    float vkx = pos[3 * k] - pos[3 * c], vky = pos[3 * k + 1] - pos[3 * c + 1], vkz = pos[3 * k + 2] - pos[3 * c + 2];
    float dj = sqrtf(vjx * vjx + vjy * vjy + vjz * vjz);
    float dk = sqrtf(vkx * vkx + vky * vky + vkz * vkz);
    float u  = (vjx * vkx + vjy * vky + vjz * vkz) / (dj * dk);
    u = fminf(1.0f, fmaxf(-1.0f, u));
    float cost = 0.95f * u;
    float th   = acosf(cost);
    float fcj = fcut(dj, RCA_), fck = fcut(dk, RCA_);
    float fc = fcj * fck;
    float davg = 0.5f * (dj + dk);
    int sj_ = sp[j], sk_ = sp[k];
    int s1 = min(sj_, sk_), s2 = max(sj_, sk_);
    int pidx = s1 * SPN - (s1 * (s1 + 1)) / 2 + s2;
    const float* gsrc = &daev[(long)c * AEVD + 112 + pidx * 32];

    float f2[8], f2p[8];
    #pragma unroll
    for (int a = 0; a < 8; ++a) {
        float q = davg - (0.8f + 0.3375f * a);
        f2[a]  = __expf(-ETAA * q * q);
        f2p[a] = -2.0f * ETAA * q * f2[a];
    }
    float S0 = 0.0f, S1 = 0.0f, S2 = 0.0f;
    #pragma unroll
    for (int z = 0; z < 4; ++z) {
        float phi = th - (z + 0.5f) * (PIF * 0.25f);
        float b   = 0.5f * (1.0f + __cosf(phi));
        float f1  = __powf(b, ZET);
        float f1p = ZET * __powf(b, ZET - 1.0f) * (-0.5f * __sinf(phi));
        #pragma unroll
        for (int a = 0; a < 8; ++a) {
            float g = gsrc[z * 8 + a];
            S0 += g * f1p * f2[a];
            S1 += g * f1  * f2p[a];
            S2 += g * f1  * f2[a];
        }
    }
    float dSdth   = 2.0f * fc * S0;
    float dSddavg = 2.0f * fc * S1;
    float A2      = 2.0f * S2;
    float dSdcos  = dSdth * (-1.0f / sqrtf(fmaxf(1.0f - cost * cost, 1e-6f)));
    float dSddj = 0.5f * dSddavg + A2 * fck * fcutp(dj, RCA_);
    float dSddk = 0.5f * dSddavg + A2 * fcj * fcutp(dk, RCA_);
    float inv_jk = 1.0f / (dj * dk);
    float cj = dSddj / dj, ck = dSddk / dk;
    float q95 = 0.95f * dSdcos;
    float gjx = cj * vjx + q95 * (vkx * inv_jk - u * vjx / (dj * dj));
    float gjy = cj * vjy + q95 * (vky * inv_jk - u * vjy / (dj * dj));
    float gjz = cj * vjz + q95 * (vkz * inv_jk - u * vjz / (dj * dj));
    float gkx = ck * vkx + q95 * (vjx * inv_jk - u * vkx / (dk * dk));
    float gky = ck * vky + q95 * (vjy * inv_jk - u * vky / (dk * dk));
    float gkz = ck * vkz + q95 * (vjz * inv_jk - u * vkz / (dk * dk));
    atomicAdd(&grad[3 * j + 0], gjx); atomicAdd(&grad[3 * j + 1], gjy); atomicAdd(&grad[3 * j + 2], gjz);
    atomicAdd(&grad[3 * k + 0], gkx); atomicAdd(&grad[3 * k + 1], gky); atomicAdd(&grad[3 * k + 2], gkz);
    atomicAdd(&grad[3 * c + 0], -(gjx + gkx));
    atomicAdd(&grad[3 * c + 1], -(gjy + gky));
    atomicAdd(&grad[3 * c + 2], -(gjz + gkz));
}

__global__ void k_finalize(const float* __restrict__ en, const float* __restrict__ grad,
                           float* __restrict__ out) {
    int i = blockIdx.x * blockDim.x + threadIdx.x;
    if (i < NA * 3) out[1 + i] = -grad[i];
    if (i == 0) out[0] = en[0];
}

// ---------------------------------------------------------------------------
// host launcher
// ---------------------------------------------------------------------------
extern "C" void kernel_launch(void* const* d_in, const int* in_sizes, int n_in,
                              void* d_out, int out_size, void* d_ws, size_t ws_size,
                              hipStream_t stream) {
    const int*   sp  = (const int*)d_in[0];
    const float* pos = (const float*)d_in[1];
    const int*   i12 = (const int*)d_in[2];
    const int P = in_sizes[2] / 2;
    const int*   tc  = (const int*)d_in[5];
    const int T = in_sizes[5];
    const int*   tj  = (const int*)d_in[6];
    const int*   tk  = (const int*)d_in[7];
    const int*   idx = (const int*)d_in[8];
    const float* W0 = (const float*)d_in[9];
    const float* b0 = (const float*)d_in[10];
    const float* W1 = (const float*)d_in[11];
    const float* b1 = (const float*)d_in[12];
    const float* W2 = (const float*)d_in[13];
    const float* b2 = (const float*)d_in[14];
    const float* W3 = (const float*)d_in[15];
    const float* b3 = (const float*)d_in[16];
    const float* sae = (const float*)d_in[17];
    float* out = (float*)d_out;

    char* wsb = (char*)d_ws;
    size_t off = 0;
    auto alloc = [&](size_t bytes) -> char* {
        char* p = wsb + off;
        off = (off + bytes + 255) & ~(size_t)255;
        return p;
    };
    float* aev  = (float*)alloc(sizeof(float) * (size_t)NA * AEVD);
    float* daev = (float*)alloc(sizeof(float) * (size_t)NA * AEVD);
    float* grad = (float*)alloc(sizeof(float) * (size_t)NA * 3);
    float* en   = (float*)alloc(sizeof(float) * 16);
    float* gx   = (float*)alloc(sizeof(float) * (size_t)SPN * MP * K0);
    _Float16* xh  = (_Float16*)alloc(2ull * SPN * MP * K0);
    _Float16* w0h = (_Float16*)alloc(2ull * SPN * EN * K0 * N1);
    _Float16* w0t = (_Float16*)alloc(2ull * SPN * EN * N1 * K0);
    _Float16* w1h = (_Float16*)alloc(2ull * SPN * EN * N1 * N2);
    _Float16* w1t = (_Float16*)alloc(2ull * SPN * EN * N2 * N1);
    _Float16* w2h = (_Float16*)alloc(2ull * SPN * EN * N2 * N3);
    _Float16* w2t = (_Float16*)alloc(2ull * SPN * EN * N3 * N2);
    _Float16* h0 = (_Float16*)alloc(2ull * SPN * EN * MP * N1);
    _Float16* d0 = (_Float16*)alloc(2ull * SPN * EN * MP * N1);
    _Float16* g0 = (_Float16*)alloc(2ull * SPN * EN * MP * N1);
    _Float16* h1 = (_Float16*)alloc(2ull * SPN * EN * MP * N2);
    _Float16* d1 = (_Float16*)alloc(2ull * SPN * EN * MP * N2);
    _Float16* g1 = (_Float16*)alloc(2ull * SPN * EN * MP * N2);
    _Float16* h2 = (_Float16*)alloc(2ull * SPN * EN * MP * N3);
    _Float16* d2 = (_Float16*)alloc(2ull * SPN * EN * MP * N3);
    _Float16* g2 = (_Float16*)alloc(2ull * SPN * EN * MP * N3);
    (void)ws_size; (void)n_in; (void)out_size;

    dim3 B256(256);
    auto G1 = [](long n) { return dim3((unsigned)((n + 255) / 256)); };

    // zero accumulators
    k_zero<<<G1((long)NA * AEVD), B256, 0, stream>>>(aev, (long)NA * AEVD);
    k_zero<<<G1((long)NA * AEVD), B256, 0, stream>>>(daev, (long)NA * AEVD);
    k_zero<<<G1(NA * 3), B256, 0, stream>>>(grad, NA * 3);
    k_zero<<<1, 1, 0, stream>>>(en, 1);

    // weight conversion (f32 -> f16 normal + transposed)
    long cw0 = (long)SPN * EN * K0 * N1;
    k_convw<<<G1(cw0), B256, 0, stream>>>(W0, w0h, w0t, AEVD, K0, N1, cw0);
    long cw1 = (long)SPN * EN * N1 * N2;
    k_convw<<<G1(cw1), B256, 0, stream>>>(W1, w1h, w1t, N1, N1, N2, cw1);
    long cw2 = (long)SPN * EN * N2 * N3;
    k_convw<<<G1(cw2), B256, 0, stream>>>(W2, w2h, w2t, N2, N2, N3, cw2);

    // AEV forward
    k_rad_fwd<<<G1(P), B256, 0, stream>>>(pos, sp, i12, P, aev);
    k_ang_fwd<<<G1(T), B256, 0, stream>>>(pos, sp, tc, tj, tk, T, aev);
    long gtot = (long)SPN * MP * K0;
    k_gather_x<<<G1(gtot), B256, 0, stream>>>(aev, idx, xh, gtot);

    // MLP forward (WMMA, 16x64 / 16x32 wave tiles)
    dim3 W32(32);
    dim3 gf0(MT * (N1 / 64), EN, SPN);
    k_gemm<0, 4><<<gf0, W32, 0, stream>>>(
        xh, (long)MP * K0, 0L, K0,
        w0t, (long)EN * N1 * K0, (long)N1 * K0, K0,
        b0, N1, nullptr, 0, 0,
        h0, (long)EN * MP * N1, (long)MP * N1, N1, d0,
        nullptr, 0, K0, N1 / 64, 1);
    dim3 gf1(MT * (N2 / 64), EN, SPN);
    k_gemm<0, 4><<<gf1, W32, 0, stream>>>(
        h0, (long)EN * MP * N1, (long)MP * N1, N1,
        w1t, (long)EN * N2 * N1, (long)N2 * N1, N1,
        b1, N2, nullptr, 0, 0,
        h1, (long)EN * MP * N2, (long)MP * N2, N2, d1,
        nullptr, 0, N1, N2 / 64, 1);
    dim3 gf2(MT * (N3 / 32), EN, SPN);
    k_gemm<0, 2><<<gf2, W32, 0, stream>>>(
        h1, (long)EN * MP * N2, (long)MP * N2, N2,
        w2t, (long)EN * N3 * N2, (long)N3 * N2, N2,
        b2, N3, nullptr, 0, 0,
        h2, (long)EN * MP * N3, (long)MP * N3, N3, d2,
        nullptr, 0, N2, N3 / 32, 1);

    // layer 3 + energy; backward seed G2
    k_l3_energy<<<G1((long)SPN * EN * MS), B256, 0, stream>>>(h2, W3, b3, sae, en);
    long g2tot = (long)SPN * EN * MP * N3;
    k_g2_seed<<<G1(g2tot), B256, 0, stream>>>(d2, W3, g2, g2tot);

    // MLP backward (WMMA)
    dim3 gb2(MT * (N2 / 64), EN, SPN);
    k_gemm<1, 4><<<gb2, W32, 0, stream>>>(
        g2, (long)EN * MP * N3, (long)MP * N3, N3,
        w2h, (long)EN * N2 * N3, (long)N2 * N3, N3,
        nullptr, 0,
        d1, (long)EN * MP * N2, (long)MP * N2,
        g1, (long)EN * MP * N2, (long)MP * N2, N2, nullptr,
        nullptr, 0, N3, N2 / 64, 1);
    dim3 gb1(MT * (N1 / 64), EN, SPN);
    k_gemm<1, 4><<<gb1, W32, 0, stream>>>(
        g1, (long)EN * MP * N2, (long)MP * N2, N2,
        w1h, (long)EN * N1 * N2, (long)N1 * N2, N2,
        nullptr, 0,
        d0, (long)EN * MP * N1, (long)MP * N1,
        g0, (long)EN * MP * N1, (long)MP * N1, N1, nullptr,
        nullptr, 0, N2, N1 / 64, 1);
    dim3 gbx(MT * (K0 / 64), 1, SPN);
    k_gemm<2, 4><<<gbx, W32, 0, stream>>>(
        g0, (long)EN * MP * N1, (long)MP * N1, N1,
        w0h, (long)EN * K0 * N1, (long)K0 * N1, N1,
        nullptr, 0, nullptr, 0, 0,
        nullptr, 0, 0, K0, nullptr,
        gx, (long)MP * K0, N1, K0 / 64, EN);

    // dAEV scatter and AEV backward to coordinates
    long stot = (long)SPN * MS * AEVD;
    k_scatter_daev<<<G1(stot), B256, 0, stream>>>(gx, idx, daev, stot);
    k_rad_bwd<<<G1(P), B256, 0, stream>>>(pos, sp, i12, P, daev, grad);
    k_ang_bwd<<<G1(T), B256, 0, stream>>>(pos, sp, tc, tj, tk, T, daev, grad);

    // outputs: energy + forces (-grad)
    k_finalize<<<G1(NA * 3), B256, 0, stream>>>(en, grad, out);
}
